// Net_sparse_89120571392624
// MI455X (gfx1250) — matrix operand order repaired and checked by reading, not compiled, not run on hardware
//
#include <hip/hip_runtime.h>

typedef __bf16 bf16;
typedef __attribute__((ext_vector_type(16))) __bf16 v16bf;
typedef __attribute__((ext_vector_type(8)))  __bf16 v8bf;
typedef __attribute__((ext_vector_type(8)))  float  v8f;

union BF16x16 { v16bf v; v8bf h[2]; };

// Async global->LDS copy of 16 bytes per active lane (CDNA5 GLOBAL_LOAD_ASYNC_TO_LDS_B128).
__device__ __forceinline__ void async_copy16(void* lds_ptr, const void* gptr) {
  unsigned lds_addr = (unsigned)(unsigned long long)lds_ptr;   // low 32 bits = LDS offset
  unsigned long long gaddr = (unsigned long long)gptr;
  asm volatile("global_load_async_to_lds_b128 %0, %1, off"
               :: "v"(lds_addr), "v"(gaddr) : "memory");
}
__device__ __forceinline__ void wait_async0() {
  asm volatile("s_wait_asynccnt 0x0" ::: "memory");
}

// ---------------------------------------------------------------------------
// Rearrange conv2 weights W2[64][32][3][3] (f32) into bf16 WMMA B-fragments.
// idx = ((kt*4 + nt)*32 + lane)*16 + j ; K = kt*32 + (lane<16? j : 16+j) ;
// N = nt*16 + (lane&15) ; B[t*32+c][o] = W2[o][c][ky][kx]
// ---------------------------------------------------------------------------
__global__ void __launch_bounds__(256) conv2w_prep(const float* __restrict__ W2,
                                                   bf16* __restrict__ wfrag) {
  int idx = blockIdx.x * 256 + threadIdx.x;
  if (idx >= 9 * 4 * 32 * 16) return;
  int j    = idx & 15;
  int lane = (idx >> 4) & 31;
  int nt   = (idx >> 9) & 3;
  int kt   = idx >> 11;
  int c  = ((lane < 16) ? 0 : 16) + j;
  int o  = nt * 16 + (lane & 15);
  int ky = kt / 3, kx = kt % 3;
  wfrag[idx] = (bf16)W2[((o * 32 + c) * 3 + ky) * 3 + kx];
}

// fc1_w[128][12544] -> bf16 B-fragments: idx = ((kt*8+nt)*32+lane)*16 + j
__global__ void __launch_bounds__(256) fc1w_prep(const float* __restrict__ fc1_w,
                                                 bf16* __restrict__ wfrag) {
  int idx = blockIdx.x * 256 + threadIdx.x;
  if (idx >= 392 * 8 * 32 * 16) return;
  int j    = idx & 15;
  int lane = (idx >> 4) & 31;
  int nt   = (idx >> 9) & 7;
  int kt   = idx >> 12;
  int K = kt * 32 + ((lane < 16) ? j : 16 + j);
  int N = nt * 16 + (lane & 15);
  wfrag[idx] = (bf16)fc1_w[(size_t)N * 12544 + K];
}

// ---------------------------------------------------------------------------
// Fused conv1(1->32)+mask+relu -> halo-padded LDS ; conv2(32->64) WMMA
// implicit GEMM (+mask+relu) ; 2x2 maxpool in-register ; pooled bf16 out in
// flatten order [B][c*196 + py*14 + px].  One block = one image, 8 waves.
// ---------------------------------------------------------------------------
__global__ void __launch_bounds__(256) conv_fused(
    const float* __restrict__ x, const unsigned char* __restrict__ mask,
    const float* __restrict__ W1, const bf16* __restrict__ w2frag,
    bf16* __restrict__ pooled)
{
  __shared__ __align__(16) float         xs[784];
  __shared__ __align__(16) unsigned char msb[784];
  __shared__ __align__(16) float         w1[288];
  __shared__ __align__(16) bf16          h1[900 * 32];  // 30x30 padded, [pos][inCh]

  const int tid = threadIdx.x;
  const int b   = blockIdx.x;

  const char* xim = (const char*)(x + (size_t)b * 784);
  const char* mim = (const char*)(mask + (size_t)b * 784);

  // ---- async-stage image (3136 B) + mask bytes (784 B) into LDS ----
  if (tid < 196) async_copy16((char*)xs + tid * 16, xim + tid * 16);
  if (tid < 49)  async_copy16((char*)msb + tid * 16, mim + tid * 16);
  if (tid < 288) w1[tid] = W1[tid];   // w1[c*9 + ky*3 + kx]

  // zero the 1-wide halo border of h1 (116 border positions x 32ch x 2B)
  if (tid < 116) {
    int r, c2;
    if      (tid < 30) { r = 0;             c2 = tid;      }
    else if (tid < 60) { r = 29;            c2 = tid - 30; }
    else if (tid < 88) { r = tid - 60 + 1;  c2 = 0;        }
    else               { r = tid - 88 + 1;  c2 = 29;       }
    uint4* z = (uint4*)(h1 + (r * 30 + c2) * 32);
    const uint4 zz = {0u, 0u, 0u, 0u};
    z[0] = zz; z[1] = zz; z[2] = zz; z[3] = zz;
  }
  wait_async0();
  __syncthreads();

  // ---- conv1 + mask + relu -> bf16 LDS (interior of padded image) ----
  for (int p = tid; p < 784; p += 256) {
    const int py = p / 28, px = p % 28;
    float xv[9];
#pragma unroll
    for (int ky = 0; ky < 3; ++ky)
#pragma unroll
      for (int kx = 0; kx < 3; ++kx) {
        const int iy = py + ky - 1, ix = px + kx - 1;
        xv[ky * 3 + kx] =
            (iy >= 0 && iy < 28 && ix >= 0 && ix < 28) ? xs[iy * 28 + ix] : 0.0f;
      }
    const bool mv = msb[p] != 0;
    bf16* hp = h1 + ((py + 1) * 30 + (px + 1)) * 32;
    for (int c = 0; c < 32; ++c) {
      float s = 0.0f;
#pragma unroll
      for (int t = 0; t < 9; ++t) s += xv[t] * w1[c * 9 + t];
      hp[c] = (bf16)(mv ? fmaxf(s, 0.0f) : 0.0f);
    }
  }
  __syncthreads();

  // ---- conv2 via WMMA: M-tile = 2x8 spatial patch, N = 64 ch, K = 288 ----
  const int wave = tid >> 5;
  const int lane = tid & 31;
  const int l15  = lane & 15;
  const bool lo  = lane < 16;
  const int cb   = lo ? 0 : 16;     // A-matrix 16x32 bf16 channel base per lane half

  for (int tt = wave; tt < 56; tt += 8) {        // 14 pool-rows x 4 col-groups
    const int py = tt >> 2;                      // pooled row 0..13
    const int xg = tt & 3;                       // 8-col group
    const int x0 = xg * 8;
    const int r0 = py * 2;
    // padded coords of this lane's A row position
    const int ayp = r0 + (l15 >> 3) + 1;         // 1..28
    const int axp = x0 + (l15 & 7) + 1;          // 1..32 (clamped per-tap)

    v8f acc[4];
#pragma unroll
    for (int nt = 0; nt < 4; ++nt) acc[nt] = (v8f){};

#pragma unroll
    for (int kt = 0; kt < 9; ++kt) {             // taps, K-step = 32 in-ch
      const int ry = ayp + kt / 3 - 1;           // 0..29, always in range
      int rx = axp + kt % 3 - 1;                 // -? ..33 -> clamp into zero halo
      rx = rx > 29 ? 29 : rx;
      const bf16* hp = h1 + (ry * 30 + rx) * 32 + cb;
      BF16x16 A;
      A.h[0] = *(const v8bf*)(hp);               // c = cb..cb+7
      A.h[1] = *(const v8bf*)(hp + 8);           // c = cb+8..cb+15
      const bf16* wb = w2frag + ((size_t)(kt * 4) * 32 + lane) * 16;
#pragma unroll
      for (int nt = 0; nt < 4; ++nt) {
        BF16x16 B;
        B.h[0] = *(const v8bf*)(wb + nt * 512);
        B.h[1] = *(const v8bf*)(wb + nt * 512 + 8);
        acc[nt] = __builtin_amdgcn_wmma_f32_16x16x32_bf16(
            false, A.v, false, B.v, (short)0, acc[nt], false, false);
      }
    }

    // ---- mask + relu + 2x2 maxpool, in-register ----
    // C layout: lane<16 -> M=j (row r0), lane>=16 -> M=j+8 (row r0+1); N=lane&15
    const int mrow = (r0 + (lo ? 0 : 1)) * 28;
#pragma unroll
    for (int nt = 0; nt < 4; ++nt) {
      float v[8];
#pragma unroll
      for (int j = 0; j < 8; ++j) {
        const int cx = x0 + j;
        v[j] = (cx < 28 && msb[mrow + cx]) ? fmaxf(acc[nt][j], 0.0f) : 0.0f;
      }
#pragma unroll
      for (int j = 0; j < 8; ++j)                 // vertical max: row r0 vs r0+1
        v[j] = fmaxf(v[j], __shfl_xor(v[j], 16, 32));
      if (lo) {
        const int ch = nt * 16 + lane;
        bf16* op = pooled + (size_t)b * 12544 + ch * 196 + py * 14 + xg * 4;
#pragma unroll
        for (int k = 0; k < 4; ++k)               // horizontal max, write pooled
          if (xg * 4 + k < 14) op[k] = (bf16)fmaxf(v[2 * k], v[2 * k + 1]);
      }
    }
  }
}

// ---------------------------------------------------------------------------
// FC1: [4096 x 12544] @ [12544 x 128] + bias, relu -> bf16.
// One block per 16-row M-tile; 8 waves = 8 N-tiles of 16; K-loop of 392 WMMAs.
// ---------------------------------------------------------------------------
__global__ void __launch_bounds__(256) fc1_gemm(
    const bf16* __restrict__ pooled, const bf16* __restrict__ wfrag,
    const float* __restrict__ fc1_b, bf16* __restrict__ hfc1)
{
  const int mtile = blockIdx.x;
  const int wave  = threadIdx.x >> 5;     // N-tile
  const int lane  = threadIdx.x & 31;
  const bool lo   = lane < 16;

  const int M = mtile * 16 + (lane & 15);
  const bf16* arow  = pooled + (size_t)M * 12544 + (lo ? 0 : 16);
  const bf16* bbase = wfrag + ((size_t)wave * 32 + lane) * 16;

  v8f acc = {};
#pragma unroll 2
  for (int kt = 0; kt < 392; ++kt) {
    BF16x16 A, B;
    const bf16* ap = arow + kt * 32;
    A.h[0] = *(const v8bf*)(ap);
    A.h[1] = *(const v8bf*)(ap + 8);
    const bf16* bp = bbase + (size_t)kt * 4096;   // 8 ntiles * 32 lanes * 16
    __builtin_prefetch(bp + 4 * 4096, 0, 1);
    B.h[0] = *(const v8bf*)(bp);
    B.h[1] = *(const v8bf*)(bp + 8);
    acc = __builtin_amdgcn_wmma_f32_16x16x32_bf16(
        false, A.v, false, B.v, (short)0, acc, false, false);
  }

  const int N = wave * 16 + (lane & 15);
  const float bias = fc1_b[N];
  const int Mb = mtile * 16 + (lo ? 0 : 8);
#pragma unroll
  for (int j = 0; j < 8; ++j) {
    const float v = fmaxf(acc[j] + bias, 0.0f);
    hfc1[(size_t)(Mb + j) * 128 + N] = (bf16)v;
  }
}

// ---------------------------------------------------------------------------
// FC2 (128->10) + bias + log_softmax ; one thread per batch row.
// ---------------------------------------------------------------------------
__global__ void __launch_bounds__(256) fc2_logsoftmax(
    const bf16* __restrict__ hfc1, const float* __restrict__ fc2_w,
    const float* __restrict__ fc2_b, float* __restrict__ out)
{
  const int b = blockIdx.x * 256 + threadIdx.x;
  if (b >= 4096) return;
  float acc[10];
#pragma unroll
  for (int o = 0; o < 10; ++o) acc[o] = fc2_b[o];
  const bf16* h = hfc1 + (size_t)b * 128;
  for (int k = 0; k < 128; ++k) {
    const float hv = (float)h[k];
#pragma unroll
    for (int o = 0; o < 10; ++o) acc[o] += hv * fc2_w[o * 128 + k];
  }
  float mx = acc[0];
#pragma unroll
  for (int o = 1; o < 10; ++o) mx = fmaxf(mx, acc[o]);
  float s = 0.0f;
#pragma unroll
  for (int o = 0; o < 10; ++o) s += __expf(acc[o] - mx);
  const float lse = mx + __logf(s);
  float* op = out + (size_t)b * 10;
#pragma unroll
  for (int o = 0; o < 10; ++o) op[o] = acc[o] - lse;
}

// ---------------------------------------------------------------------------
extern "C" void kernel_launch(void* const* d_in, const int* in_sizes, int n_in,
                              void* d_out, int out_size, void* d_ws, size_t ws_size,
                              hipStream_t stream) {
  const float*         x     = (const float*)d_in[0];
  const unsigned char* mask  = (const unsigned char*)d_in[1];   // bool array
  const float*         W1    = (const float*)d_in[2];
  const float*         W2    = (const float*)d_in[3];
  const float*         fc1_w = (const float*)d_in[4];
  const float*         fc1_b = (const float*)d_in[5];
  const float*         fc2_w = (const float*)d_in[6];
  const float*         fc2_b = (const float*)d_in[7];
  float* out = (float*)d_out;

  char* ws = (char*)d_ws;
  bf16* pooled  = (bf16*)(ws);                                   // 102,760,448 B
  bf16* w2frag  = (bf16*)(ws + 102760448);                       //      36,864 B
  bf16* fc1frag = (bf16*)(ws + 102760448 + 36864);               //   3,211,264 B
  bf16* hfc1    = (bf16*)(ws + 102760448 + 36864 + 3211264);     //   1,048,576 B

  conv2w_prep   <<<(18432 + 255) / 256,   256, 0, stream>>>(W2, w2frag);
  fc1w_prep     <<<(1605632 + 255) / 256, 256, 0, stream>>>(fc1_w, fc1frag);
  conv_fused    <<<4096, 256, 0, stream>>>(x, mask, W1, w2frag, pooled);
  fc1_gemm      <<<256,  256, 0, stream>>>(pooled, fc1frag, fc1_b, hfc1);
  fc2_logsoftmax<<<16,   256, 0, stream>>>(hfc1, fc2_w, fc2_b, out);
}
